// VQEmbedding_15977278341364
// MI455X (gfx1250) — matrix operand order repaired and checked by reading, not compiled, not run on hardware
//
#include <hip/hip_runtime.h>
#include <stdint.h>

// CDNA5 / gfx1250: wave32, WMMA 16x16x32 f16 -> f32 accum,
// async global->LDS copies (ASYNCcnt) for codebook staging.
typedef __attribute__((ext_vector_type(16))) _Float16 v16h;
typedef __attribute__((ext_vector_type(8)))  float    v8f;

#define VQ_D      256   // embedding dim
#define VQ_MTILE  128   // rows per workgroup (8 waves x 16 rows)
#define VQ_NCHUNK 16    // codes per chunk (one 16x16 WMMA tile of distances)

#define VQ_WAIT_ASYNC() asm volatile("s_wait_asynccnt 0" ::: "memory")

// ---------------------------------------------------------------------------
// Prep: codebook f32 -> f16 into workspace, plus per-code squared L2 norms.
// ---------------------------------------------------------------------------
__global__ __launch_bounds__(256) void vq_prep_kernel(
    const float* __restrict__ cb, _Float16* __restrict__ cb16,
    float* __restrict__ norms) {
  const int k = blockIdx.x;
  const int t = threadIdx.x;
  const float v = cb[(size_t)k * VQ_D + t];
  cb16[(size_t)k * VQ_D + t] = (_Float16)v;

  __shared__ float red[256];
  red[t] = v * v;
  __syncthreads();
#pragma unroll
  for (int s = 128; s > 0; s >>= 1) {
    if (t < s) red[t] += red[t + s];
    __syncthreads();
  }
  if (t == 0) norms[k] = red[0];
}

// ---------------------------------------------------------------------------
// Main kernel.
// ---------------------------------------------------------------------------
__global__ __launch_bounds__(256) void vq_main_kernel(
    const float* __restrict__ z_e, const float* __restrict__ cbf32,
    const _Float16* __restrict__ cb16, const float* __restrict__ norms,
    float* __restrict__ out, int nRows, int K) {
  // Double-buffered codebook chunk (16 codes x 256 dims, f16) + result indices.
  __shared__ alignas(64) _Float16 sB[2][VQ_NCHUNK * VQ_D];  // 2 x 8 KB
  __shared__ int sIdx[VQ_MTILE];

  const int tid  = threadIdx.x;
  const int lane = tid & 31;
  const int wave = tid >> 5;
  const int h    = lane >> 4;   // lane half (0: lanes 0-15, 1: lanes 16-31)
  const int mrow = lane & 15;   // A row / B,C column within the 16x16 tile
  const int wgBase = blockIdx.x * VQ_MTILE;
  const int grow   = wgBase + wave * 16 + mrow;

  // ---- Preload + convert A fragments (stationary in VGPRs) ----
  // A 16x32 in contiguous k order (B matches): half 0 holds K = {kk..kk+15},
  //                        half 1 holds K = {kk+16..kk+31}.
  const float* zr = z_e + (size_t)grow * VQ_D;
  v16h afrag[8];
#pragma unroll
  for (int f = 0; f < 8; ++f) {
    const int kk = f * 32;
    const float4* p0 = (const float4*)(zr + kk + 16 * h);
    const float4 a0 = p0[0], a1 = p0[1];
    const float4 a2 = p0[2], a3 = p0[3];
    v16h a;
    a[0]  = (_Float16)a0.x; a[1]  = (_Float16)a0.y;
    a[2]  = (_Float16)a0.z; a[3]  = (_Float16)a0.w;
    a[4]  = (_Float16)a1.x; a[5]  = (_Float16)a1.y;
    a[6]  = (_Float16)a1.z; a[7]  = (_Float16)a1.w;
    a[8]  = (_Float16)a2.x; a[9]  = (_Float16)a2.y;
    a[10] = (_Float16)a2.z; a[11] = (_Float16)a2.w;
    a[12] = (_Float16)a3.x; a[13] = (_Float16)a3.y;
    a[14] = (_Float16)a3.z; a[15] = (_Float16)a3.w;
    afrag[f] = a;
  }

  // Async copy of one 8 KB codebook chunk into LDS buffer `buf`:
  // 256 threads x 2 segments x 16 B, tracked by ASYNCcnt.
  auto asyncCopyChunk = [&](int cbase, int buf) {
    const char* gsrc = (const char*)(cb16 + (size_t)cbase * VQ_D);
    const uint32_t ldsBase = (uint32_t)(uintptr_t)&sB[buf][0];
#pragma unroll
    for (int s = 0; s < 2; ++s) {
      const int seg = tid + s * 256;
      const uint32_t loff = ldsBase + seg * 16;
      const char* g = gsrc + seg * 16;
      asm volatile("global_load_async_to_lds_b128 %0, %1, off"
                   :: "v"(loff), "v"(g) : "memory");
    }
  };

  // Per-lane running argmin over this lane's own N column.
  float bestVal[8];
  int   bestIdx[8];
#pragma unroll
  for (int r = 0; r < 8; ++r) { bestVal[r] = 3.402823466e38f; bestIdx[r] = 0; }

  const int nChunks = K / VQ_NCHUNK;

  // Prologue: stage chunk 0.
  asyncCopyChunk(0, 0);
  VQ_WAIT_ASYNC();
  __syncthreads();

#pragma unroll 1
  for (int c = 0; c < nChunks; ++c) {
    const int cbase = c * VQ_NCHUNK;
    // Prefetch next chunk into the other buffer while we compute.
    if (c + 1 < nChunks) asyncCopyChunk(cbase + VQ_NCHUNK, (c + 1) & 1);

    // B fragments from LDS: lane holds column n = mrow, 16 contiguous K
    // values starting at f*32 + 16*h (contiguous k order, matches A).
    const _Float16* bp = &sB[c & 1][mrow * VQ_D + 16 * h];
    v16h b[8];
#pragma unroll
    for (int f = 0; f < 8; ++f) b[f] = *(const v16h*)(bp + f * 32);

    v8f acc = {};
#pragma unroll
    for (int f = 0; f < 8; ++f) {
      acc = __builtin_amdgcn_wmma_f32_16x16x32_f16(
          /*neg_a=*/false, afrag[f], /*neg_b=*/false, b[f],
          /*c_mod=*/(short)0, acc, /*reuse_a=*/false, /*reuse_b=*/false);
    }

    const int n = cbase + mrow;
    const float nc = norms[n];
#pragma unroll
    for (int r = 0; r < 8; ++r) {
      // C/D layout: lane half h, VGPR r -> (M = r + 8h, N = mrow).
      const float v = nc - 2.0f * acc[r];
      if (v < bestVal[r]) { bestVal[r] = v; bestIdx[r] = n; }
    }

    VQ_WAIT_ASYNC();   // next chunk's LDS data landed
    __syncthreads();   // all waves done reading sB[c&1] before it is reused
  }

  // ---- Final cross-lane argmin within each 16-lane group ----
#pragma unroll
  for (int r = 0; r < 8; ++r) {
    float v  = bestVal[r];
    int   ix = bestIdx[r];
#pragma unroll
    for (int m = 8; m >= 1; m >>= 1) {
      const float ov = __shfl_xor(v, m, 16);
      const int   oi = __shfl_xor(ix, m, 16);
      if (ov < v || (ov == v && oi < ix)) { v = ov; ix = oi; }
    }
    bestIdx[r] = ix;
  }
  if (mrow == 0) {
#pragma unroll
    for (int r = 0; r < 8; ++r)
      sIdx[wave * 16 + 8 * h + r] = bestIdx[r];
  }
  __syncthreads();

  // ---- Gather f32 codes, write z_q and z_q_bar (identical numerics) ----
  const float4* cb4 = (const float4*)cbf32;
  float4* o4 = (float4*)out;
  const size_t outStride = (size_t)nRows * (VQ_D / 4);
#pragma unroll 1
  for (int i = tid; i < VQ_MTILE * (VQ_D / 4); i += 256) {
    const int row = i >> 6;            // VQ_D/4 == 64
    const int c4  = i & 63;
    const int idx = sIdx[row];
    const float4 v = cb4[(size_t)idx * 64 + c4];
    const size_t off = (size_t)(wgBase + row) * 64 + c4;
    o4[off]             = v;
    o4[outStride + off] = v;
  }
}

// ---------------------------------------------------------------------------
extern "C" void kernel_launch(void* const* d_in, const int* in_sizes, int n_in,
                              void* d_out, int out_size, void* d_ws,
                              size_t ws_size, hipStream_t stream) {
  const float* z_e = (const float*)d_in[0];   // [B*S, D] f32
  const float* cb  = (const float*)d_in[1];   // [K, D]   f32
  const int nRows = in_sizes[0] / VQ_D;       // 65536
  const int K     = in_sizes[1] / VQ_D;       // 1024

  // Workspace layout: [K*D f16 codebook][K f32 norms]
  _Float16* cb16 = (_Float16*)d_ws;
  float* norms = (float*)((char*)d_ws + (size_t)K * VQ_D * sizeof(_Float16));

  vq_prep_kernel<<<K, 256, 0, stream>>>(cb, cb16, norms);
  vq_main_kernel<<<nRows / VQ_MTILE, 256, 0, stream>>>(
      z_e, cb, cb16, norms, (float*)d_out, nRows, K);
}
